// HeterogeneousEdgeGraphSAGE_44444321579084
// MI455X (gfx1250) — compile-verified, matrix operand order checked by reading, not run on hardware
//
#include <hip/hip_runtime.h>
#include <hip/hip_bf16.h>
#include <cstring>

// ---------------------------------------------------------------------------
// CDNA5 (gfx1250, wave32) WMMA implementation of heterogeneous edge GraphSAGE.
// All dense math: v_wmma_f32_16x16x32_f16 (f16 in, f32 accumulate).
// Weights pre-converted to f16 once -> W fragments are two raw b128 loads,
// no VALU between WMMAs (kills the WMMA->VALU WAR hazard NOPs).
// Edge head stores activations in f16 (halves the HBM-bound traffic).
// ---------------------------------------------------------------------------

typedef __attribute__((ext_vector_type(16))) _Float16 v16h;
typedef __attribute__((ext_vector_type(8)))  _Float16 v8h;
typedef __attribute__((ext_vector_type(4)))  _Float16 v4h;
typedef __attribute__((ext_vector_type(8)))  float    v8f;

#define NS_N 50000
#define NT_N 50000
#define E_N  500000
#define H_N  128

struct GPair {
    const void* A;         // f32 or f16 (AHALF), [rows x lda]
    const void* W;         // f16, [>=N rows x ldw] (pre-converted torch weight)
    const float* rscale;   // optional per-row scale (1/cnt), conv only (f32 A)
    const int*   ridx;     // optional row gather, gate only
    int lda, ldw, K, pad;  // elements; K multiple of 32
};

struct GemmArgs {
    GPair p[4];
    const float* bias0;
    const float* bias1;    // nullable (fused relation pair)
    const float* resid;    // nullable f32, same layout as C
    const float* gatevec;  // gate: gw2 [N]
    const float* gatebias; // gate: gb2 [1]
    float*       gateout;  // gate: [M]
    void*        C;        // f32 or f16 (OHALF)
    int   np, M, N, ldc;
    float scale;
    int   relu;
};

// Fragment layout (ISA 7.12.2, 16-bit A/B): lane&15 = M (or N) index,
// halves 0..7 <- k0+8g .. +7, halves 8..15 <- k0+16+8g .. +7.
// f32 source: 4x global_load_b128 + 8x v_cvt_pk_f16_f32
__device__ __forceinline__ v16h load_fragF(const float* p, float s) {
    const float4 a = ((const float4*)p)[0];
    const float4 b = ((const float4*)p)[1];
    const float4 c = ((const float4*)p)[4];
    const float4 d = ((const float4*)p)[5];
    v16h f;
    f[0]  = (_Float16)(s * a.x); f[1]  = (_Float16)(s * a.y);
    f[2]  = (_Float16)(s * a.z); f[3]  = (_Float16)(s * a.w);
    f[4]  = (_Float16)(s * b.x); f[5]  = (_Float16)(s * b.y);
    f[6]  = (_Float16)(s * b.z); f[7]  = (_Float16)(s * b.w);
    f[8]  = (_Float16)(s * c.x); f[9]  = (_Float16)(s * c.y);
    f[10] = (_Float16)(s * c.z); f[11] = (_Float16)(s * c.w);
    f[12] = (_Float16)(s * d.x); f[13] = (_Float16)(s * d.y);
    f[14] = (_Float16)(s * d.z); f[15] = (_Float16)(s * d.w);
    return f;
}

// f16 source: two raw b128 loads, zero VALU
__device__ __forceinline__ v16h load_fragH(const _Float16* p) {
    v8h a = *(const v8h*)p;
    v8h b = *(const v8h*)(p + 16);
    return __builtin_shufflevector(a, b, 0, 1, 2, 3, 4, 5, 6, 7,
                                   8, 9, 10, 11, 12, 13, 14, 15);
}

// One wave owns one 16-row M-tile and the full N strip (NT16 n-tiles <= 8).
// A fragment loaded once per (pair, k-chunk), reused across all n-tiles.
template <int NT16, bool GATHER, bool SCALE, bool GATE, bool AHALF, bool OHALF>
__global__ void __launch_bounds__(128)
gemm_wmma(GemmArgs a) {
    const int lane  = threadIdx.x & 31;
    const int wave  = threadIdx.x >> 5;
    const int mtile = blockIdx.x * 4 + wave;
    if (mtile >= (a.M >> 4)) return;           // whole-wave exit, EXEC stays full
    const int mbase = mtile << 4;
    const int nlo   = lane & 15;
    const int g8    = (lane >> 4) << 3;        // +8 K-offset for upper half-wave

    v8f acc[NT16];
#pragma unroll
    for (int i = 0; i < NT16; ++i)
#pragma unroll
        for (int j = 0; j < 8; ++j) acc[i][j] = 0.0f;

#pragma unroll
    for (int ip = 0; ip < 4; ++ip) {
        if (ip >= a.np) break;
        const GPair g = a.p[ip];
        // row / scale hoisted out of the K loop (constant per pair)
        long long row = mbase + nlo;
        if (GATHER) { if (g.ridx) row = g.ridx[mbase + nlo]; }
        float s = 1.0f;
        if (SCALE) { if (g.rscale) s = g.rscale[row]; }
        const float*    pAf = (const float*)g.A    + row * (long long)g.lda + g8;
        const _Float16* pAh = (const _Float16*)g.A + row * (long long)g.lda + g8;
        const _Float16* pW  = (const _Float16*)g.W + (long long)nlo * g.ldw + g8;
#pragma unroll 2
        for (int k0 = 0; k0 < g.K; k0 += 32) {
            const v16h af = AHALF ? load_fragH(pAh + k0) : load_fragF(pAf + k0, s);
#pragma unroll
            for (int nt = 0; nt < NT16; ++nt) {
                const v16h wf = load_fragH(pW + (long long)(nt * 16) * g.ldw + k0);
                acc[nt] = __builtin_amdgcn_wmma_f32_16x16x32_f16(
                    false, af, false, wf, (short)0, acc[nt], false, false);
            }
        }
    }

    const int g16 = lane >> 4;                 // C/D rows: mbase + r + 8*g16
    if (GATE) {
        // fused gate head: gate[row] = sigmoid( relu(t) . gatevec + gatebias )
        float rs[8];
#pragma unroll
        for (int r = 0; r < 8; ++r) rs[r] = 0.0f;
#pragma unroll
        for (int nt = 0; nt < NT16; ++nt) {
            int col = (nt << 4) + nlo;
            float b  = a.bias0[col];
            float gv = a.gatevec[col];
#pragma unroll
            for (int r = 0; r < 8; ++r) {
                float t = fmaxf(acc[nt][r] + b, 0.0f);
                rs[r] += t * gv;
            }
        }
#pragma unroll
        for (int m = 1; m < 16; m <<= 1)
#pragma unroll
            for (int r = 0; r < 8; ++r) rs[r] += __shfl_xor(rs[r], m, 32);
        if (nlo == 0) {
            float gb = a.gatebias[0];
#pragma unroll
            for (int r = 0; r < 8; ++r) {
                int row = mbase + r + (g16 << 3);
                a.gateout[row] = 1.0f / (1.0f + __expf(-(rs[r] + gb)));
            }
        }
    } else {
#pragma unroll
        for (int nt = 0; nt < NT16; ++nt) {
            int col = (nt << 4) + nlo;
            if (NT16 > 1 || col < a.N) {
                float b = a.bias0[col] + (a.bias1 ? a.bias1[col] : 0.0f);
#pragma unroll
                for (int r = 0; r < 8; ++r) {
                    long long row = mbase + r + (g16 << 3);
                    float t = (acc[nt][r] + b) * a.scale;
                    if (a.resid) t += a.resid[row * a.ldc + col];
                    if (a.relu)  t = fmaxf(t, 0.0f);
                    if (OHALF) ((_Float16*)a.C)[row * a.ldc + col] = (_Float16)t;
                    else       ((float*)a.C)[row * a.ldc + col] = t;
                }
            }
        }
    }
}

// zero-padding copy (f32 out)
__global__ void pad_kernel(const float* __restrict__ src, int sr, int sc,
                           float* __restrict__ dst, int dr, int dc) {
    long long i = (long long)blockIdx.x * 256 + threadIdx.x;
    if (i >= (long long)dr * dc) return;
    int r = (int)(i / dc), c = (int)(i % dc);
    dst[i] = (r < sr && c < sc) ? src[(long long)r * sc + c] : 0.0f;
}

// zero-padding copy with f32 -> f16 convert
__global__ void pad_cvt_kernel(const float* __restrict__ src, int sr, int sc,
                               _Float16* __restrict__ dst, int dr, int dc) {
    long long i = (long long)blockIdx.x * 256 + threadIdx.x;
    if (i >= (long long)dr * dc) return;
    int r = (int)(i / dc), c = (int)(i % dc);
    dst[i] = (r < sr && c < sc) ? (_Float16)src[(long long)r * sc + c] : (_Float16)0.0f;
}

// bulk f32 -> f16 convert
__global__ void cvt_kernel(const float* __restrict__ src, _Float16* __restrict__ dst,
                           long long n) {
    long long i = (long long)blockIdx.x * 256 + threadIdx.x;
    if (i < n) dst[i] = (_Float16)src[i];
}

// scatter-add: one wave per edge, float4 per lane (32*4 = 128 = H).
__global__ void __launch_bounds__(256)
scatter_add_kernel(const float* __restrict__ x, const int* __restrict__ eidx,
                   float* __restrict__ out, int E) {
    int e = blockIdx.x * 8 + (threadIdx.x >> 5);
    if (e >= E) return;
    int lane = threadIdx.x & 31;
    int r = eidx[e];
    int c = eidx[E + e];
    float4 v = ((const float4*)(x + (long long)r * H_N))[lane];
    float* dst = out + (long long)c * H_N + lane * 4;
    atomicAdd(dst + 0, v.x);
    atomicAdd(dst + 1, v.y);
    atomicAdd(dst + 2, v.z);
    atomicAdd(dst + 3, v.w);
}

__global__ void count_kernel(const int* __restrict__ eidx, float* __restrict__ cnt, int E) {
    int e = blockIdx.x * blockDim.x + threadIdx.x;
    if (e < E) atomicAdd(cnt + eidx[E + e], 1.0f);
}

__global__ void inv_kernel(float* __restrict__ cnt, int n) {
    int i = blockIdx.x * blockDim.x + threadIdx.x;
    if (i < n) cnt[i] = 1.0f / fmaxf(cnt[i], 1.0f);
}

// combined(f16) = h3s[row]*gate + h3t[col]*(1-gate); one wave per edge.
__global__ void __launch_bounds__(256)
combine_kernel(const _Float16* __restrict__ h3s, const _Float16* __restrict__ h3t,
               const int* __restrict__ ei, const float* __restrict__ gate,
               _Float16* __restrict__ out, int E) {
    int e = blockIdx.x * 8 + (threadIdx.x >> 5);
    if (e >= E) return;
    int lane = threadIdx.x & 31;
    int r = ei[e], c = ei[E + e];
    float gv = gate[e];
    v4h s = ((const v4h*)(h3s + (long long)r * H_N))[lane];
    v4h t = ((const v4h*)(h3t + (long long)c * H_N))[lane];
    v4h o;
#pragma unroll
    for (int i = 0; i < 4; ++i)
        o[i] = (_Float16)((float)s[i] * gv + (float)t[i] * (1.0f - gv));
    ((v4h*)(out + (long long)e * H_N))[lane] = o;
}

// ---------------------------------------------------------------------------
static inline GPair mkpair(const void* A, int lda, const void* W, int ldw, int K,
                           const float* rscale = nullptr, const int* ridx = nullptr) {
    GPair g; g.A = A; g.W = W; g.rscale = rscale; g.ridx = ridx;
    g.lda = lda; g.ldw = ldw; g.K = K; g.pad = 0; return g;
}

enum GemmKind { K_ENC, K_CONV, K_GATE, K_MLP128, K_MLP64, K_MLP2OUT };

static void launch_gemm(const GemmArgs& a, GemmKind kind, hipStream_t s) {
    int blocks = ((a.M >> 4) + 3) >> 2;       // 4 waves per 128-thread block
    switch (kind) {
    case K_ENC:     gemm_wmma<8, false, false, false, false, false><<<blocks, 128, 0, s>>>(a); break;
    case K_CONV:    gemm_wmma<8, false, true,  false, false, false><<<blocks, 128, 0, s>>>(a); break;
    case K_GATE:    gemm_wmma<8, true,  false, true,  true,  false><<<blocks, 128, 0, s>>>(a); break;
    case K_MLP128:  gemm_wmma<8, false, false, false, true,  true ><<<blocks, 128, 0, s>>>(a); break;
    case K_MLP64:   gemm_wmma<4, false, false, false, true,  true ><<<blocks, 128, 0, s>>>(a); break;
    case K_MLP2OUT: gemm_wmma<1, false, false, false, true,  false><<<blocks, 128, 0, s>>>(a); break;
    }
}

extern "C" void kernel_launch(void* const* d_in, const int* in_sizes, int n_in,
                              void* d_out, int out_size, void* d_ws, size_t ws_size,
                              hipStream_t stream) {
    const float* x_source  = (const float*)d_in[0];
    const float* x_target  = (const float*)d_in[1];
    const float* edge_attr = (const float*)d_in[2];
    const float* enc_s_w1  = (const float*)d_in[3];
    const float* enc_s_b1  = (const float*)d_in[4];
    const float* enc_s_w2  = (const float*)d_in[5];
    const float* enc_s_b2  = (const float*)d_in[6];
    const float* enc_t_w1  = (const float*)d_in[7];
    const float* enc_t_b1  = (const float*)d_in[8];
    const float* enc_t_w2  = (const float*)d_in[9];
    const float* enc_t_b2  = (const float*)d_in[10];
    const float* conv_wl   = (const float*)d_in[11];   // [3,4,H,H]
    const float* conv_bl   = (const float*)d_in[12];   // [3,4,H]
    const float* conv_wr   = (const float*)d_in[13];   // [3,4,H,H]
    const float* gate_w1   = (const float*)d_in[14];   // [H,2H]
    const float* gate_b1   = (const float*)d_in[15];
    const float* gate_w2   = (const float*)d_in[16];   // [1,H]
    const float* gate_b2   = (const float*)d_in[17];
    const float* mlp_w1    = (const float*)d_in[18];   // [H,H+32]
    const float* mlp_b1    = (const float*)d_in[19];
    const float* mlp_w2    = (const float*)d_in[20];   // [64,H]
    const float* mlp_b2    = (const float*)d_in[21];
    const float* mlp_w3    = (const float*)d_in[22];   // [2,64]
    const float* mlp_b3    = (const float*)d_in[23];
    const int*   ei_ss     = (const int*)d_in[24];
    const int*   ei_tt     = (const int*)d_in[25];
    const int*   ei_st     = (const int*)d_in[26];
    const int*   ei_ts     = (const int*)d_in[27];
    (void)in_sizes; (void)n_in; (void)out_size; (void)ws_size;

    const int NS = NS_N, NT = NT_N, E = E_N, H = H_N, SF = 64, TF = 48, ED = 32;

    // ---- workspace layout (bytes, 256-aligned blocks) --------------------
    char* base = (char*)d_ws;
    size_t off = 0;
    auto allocB = [&](size_t bytes) {
        char* p = base + off;
        off += (bytes + 255) & ~(size_t)255;
        return p;
    };
    auto allocF = [&](size_t n) { return (float*)allocB(n * 4); };
    auto allocH = [&](size_t n) { return (_Float16*)allocB(n * 2); };

    float* bufA_s  = allocF((size_t)NS * H);
    float* bufB_s  = allocF((size_t)NS * H);
    float* bufA_t  = allocF((size_t)NT * H);
    float* bufB_t  = allocF((size_t)NT * H);
    float* sum_ss  = allocF((size_t)NS * H);   // also encoder tmp (source)
    float* sum_ts  = allocF((size_t)NS * H);
    float* sum_tt  = allocF((size_t)NT * H);   // also encoder tmp (target)
    float* sum_st  = allocF((size_t)NT * H);
    float* inv_ss  = allocF(NS);
    float* inv_ts  = allocF(NS);
    float* inv_tt  = allocF(NT);
    float* inv_st  = allocF(NT);
    float* gate    = allocF(E);
    float* xt_pad  = allocF((size_t)NT * 64);  // x_target padded 48 -> 64 (f32 A)
    // f16 weight copies
    _Float16* w_s1h  = allocH((size_t)H * SF);      // enc_s_w1
    _Float16* w_s2h  = allocH((size_t)H * H);       // enc_s_w2
    _Float16* w_t1h  = allocH((size_t)H * 64);      // enc_t_w1 padded [H,48]->[H,64]
    _Float16* w_t2h  = allocH((size_t)H * H);       // enc_t_w2
    _Float16* w_clh  = allocH((size_t)12 * H * H);  // conv_wl blob
    _Float16* w_crh  = allocH((size_t)12 * H * H);  // conv_wr blob
    _Float16* w_g1h  = allocH((size_t)H * 2 * H);   // gate_w1
    _Float16* w_m1h  = allocH((size_t)H * (H + ED));// mlp_w1
    _Float16* w_m2h  = allocH((size_t)64 * H);      // mlp_w2
    _Float16* w_m3h  = allocH((size_t)16 * 64);     // mlp_w3 padded [2,64]->[16,64]
    // f16 activations (edge head)
    _Float16* ea_h   = allocH((size_t)E * ED);      // edge_attr
    _Float16* h3s_h  = allocH((size_t)NS * H);
    _Float16* h3t_h  = allocH((size_t)NT * H);
    _Float16* bufE1h = allocH((size_t)E * H);
    _Float16* bufE2h = allocH((size_t)E * H);

    const int eb = (E + 255) / 256;           // 1 thread / edge
    const int ew = (E + 7) / 8;               // 1 wave / edge (256-thr blocks)
    auto grid1 = [](long long n) { return (int)((n + 255) / 256); };

    // ---- one-time prep: padding + f16 weight conversion ------------------
    pad_kernel<<<grid1((long long)NT * 64), 256, 0, stream>>>(x_target, NT, TF, xt_pad, NT, 64);
    pad_cvt_kernel<<<grid1(H * 64), 256, 0, stream>>>(enc_t_w1, H, TF, w_t1h, H, 64);
    pad_cvt_kernel<<<grid1(16 * 64), 256, 0, stream>>>(mlp_w3, 2, 64, w_m3h, 16, 64);
    cvt_kernel<<<grid1((long long)H * SF), 256, 0, stream>>>(enc_s_w1, w_s1h, (long long)H * SF);
    cvt_kernel<<<grid1((long long)H * H), 256, 0, stream>>>(enc_s_w2, w_s2h, (long long)H * H);
    cvt_kernel<<<grid1((long long)H * H), 256, 0, stream>>>(enc_t_w2, w_t2h, (long long)H * H);
    cvt_kernel<<<grid1((long long)12 * H * H), 256, 0, stream>>>(conv_wl, w_clh, (long long)12 * H * H);
    cvt_kernel<<<grid1((long long)12 * H * H), 256, 0, stream>>>(conv_wr, w_crh, (long long)12 * H * H);
    cvt_kernel<<<grid1((long long)H * 2 * H), 256, 0, stream>>>(gate_w1, w_g1h, (long long)H * 2 * H);
    cvt_kernel<<<grid1((long long)H * (H + ED)), 256, 0, stream>>>(mlp_w1, w_m1h, (long long)H * (H + ED));
    cvt_kernel<<<grid1((long long)64 * H), 256, 0, stream>>>(mlp_w2, w_m2h, (long long)64 * H);
    cvt_kernel<<<grid1((long long)E * ED), 256, 0, stream>>>(edge_attr, ea_h, (long long)E * ED);

    // ---- edge counts -> inverse means (topology is layer-invariant) -----
    hipMemsetAsync(inv_ss, 0, NS * sizeof(float), stream);
    hipMemsetAsync(inv_ts, 0, NS * sizeof(float), stream);
    hipMemsetAsync(inv_tt, 0, NT * sizeof(float), stream);
    hipMemsetAsync(inv_st, 0, NT * sizeof(float), stream);
    count_kernel<<<eb, 256, 0, stream>>>(ei_ss, inv_ss, E);
    count_kernel<<<eb, 256, 0, stream>>>(ei_ts, inv_ts, E);
    count_kernel<<<eb, 256, 0, stream>>>(ei_tt, inv_tt, E);
    count_kernel<<<eb, 256, 0, stream>>>(ei_st, inv_st, E);
    inv_kernel<<<grid1(NS), 256, 0, stream>>>(inv_ss, NS);
    inv_kernel<<<grid1(NS), 256, 0, stream>>>(inv_ts, NS);
    inv_kernel<<<grid1(NT), 256, 0, stream>>>(inv_tt, NT);
    inv_kernel<<<grid1(NT), 256, 0, stream>>>(inv_st, NT);

    // ---- node encoders ---------------------------------------------------
    {
        GemmArgs a; memset(&a, 0, sizeof a);
        a.np = 1; a.scale = 1.0f; a.M = NS; a.N = H; a.ldc = H;
        a.p[0] = mkpair(x_source, SF, w_s1h, SF, SF);
        a.bias0 = enc_s_b1; a.relu = 1; a.C = sum_ss;
        launch_gemm(a, K_ENC, stream);
        a.p[0] = mkpair(sum_ss, H, w_s2h, H, H);
        a.bias0 = enc_s_b2; a.relu = 0; a.C = bufA_s;
        launch_gemm(a, K_ENC, stream);

        a.M = NT;
        a.p[0] = mkpair(xt_pad, 64, w_t1h, 64, 64);
        a.bias0 = enc_t_b1; a.relu = 1; a.C = sum_tt;
        launch_gemm(a, K_ENC, stream);
        a.p[0] = mkpair(sum_tt, H, w_t2h, H, H);
        a.bias0 = enc_t_b2; a.relu = 0; a.C = bufA_t;
        launch_gemm(a, K_ENC, stream);
    }

    // ---- 3 hetero SAGE layers -------------------------------------------
    float* cs = bufA_s; float* ct = bufA_t;
    float* os = bufB_s; float* ot = bufB_t;
    for (int l = 0; l < 3; ++l) {
        hipMemsetAsync(sum_ss, 0, (size_t)NS * H * sizeof(float), stream);
        hipMemsetAsync(sum_ts, 0, (size_t)NS * H * sizeof(float), stream);
        hipMemsetAsync(sum_tt, 0, (size_t)NT * H * sizeof(float), stream);
        hipMemsetAsync(sum_st, 0, (size_t)NT * H * sizeof(float), stream);
        scatter_add_kernel<<<ew, 256, 0, stream>>>(cs, ei_ss, sum_ss, E);
        scatter_add_kernel<<<ew, 256, 0, stream>>>(ct, ei_tt, sum_tt, E);
        scatter_add_kernel<<<ew, 256, 0, stream>>>(cs, ei_st, sum_st, E);
        scatter_add_kernel<<<ew, 256, 0, stream>>>(ct, ei_ts, sum_ts, E);

        // dest = source nodes: relations ss (r=0) and ts (r=3)
        {
            GemmArgs a; memset(&a, 0, sizeof a);
            a.np = 4; a.scale = 0.5f; a.relu = 1;
            a.M = NS; a.N = H; a.ldc = H; a.C = os;
            a.p[0] = mkpair(sum_ss, H, w_clh + (size_t)(l * 4 + 0) * H * H, H, H, inv_ss);
            a.p[1] = mkpair(sum_ts, H, w_clh + (size_t)(l * 4 + 3) * H * H, H, H, inv_ts);
            a.p[2] = mkpair(cs, H, w_crh + (size_t)(l * 4 + 0) * H * H, H, H);
            a.p[3] = mkpair(cs, H, w_crh + (size_t)(l * 4 + 3) * H * H, H, H);
            a.bias0 = conv_bl + (size_t)(l * 4 + 0) * H;
            a.bias1 = conv_bl + (size_t)(l * 4 + 3) * H;
            a.resid = (l == 0) ? nullptr : cs;
            launch_gemm(a, K_CONV, stream);
        }
        // dest = target nodes: relations tt (r=1) and st (r=2)
        {
            GemmArgs a; memset(&a, 0, sizeof a);
            a.np = 4; a.scale = 0.5f; a.relu = 1;
            a.M = NT; a.N = H; a.ldc = H; a.C = ot;
            a.p[0] = mkpair(sum_tt, H, w_clh + (size_t)(l * 4 + 1) * H * H, H, H, inv_tt);
            a.p[1] = mkpair(sum_st, H, w_clh + (size_t)(l * 4 + 2) * H * H, H, H, inv_st);
            a.p[2] = mkpair(ct, H, w_crh + (size_t)(l * 4 + 1) * H * H, H, H);
            a.p[3] = mkpair(ct, H, w_crh + (size_t)(l * 4 + 2) * H * H, H, H);
            a.bias0 = conv_bl + (size_t)(l * 4 + 1) * H;
            a.bias1 = conv_bl + (size_t)(l * 4 + 2) * H;
            a.resid = (l == 0) ? nullptr : ct;
            launch_gemm(a, K_CONV, stream);
        }
        float* t;
        t = cs; cs = os; os = t;
        t = ct; ct = ot; ot = t;
    }
    // cs = h3s, ct = h3t (f32); make f16 copies for the edge head
    cvt_kernel<<<grid1((long long)NS * H), 256, 0, stream>>>(cs, h3s_h, (long long)NS * H);
    cvt_kernel<<<grid1((long long)NT * H), 256, 0, stream>>>(ct, h3t_h, (long long)NT * H);

    // ---- edge head -------------------------------------------------------
    // gate: fused gathered GEMM + dot + sigmoid (no E x H intermediate)
    {
        GemmArgs a; memset(&a, 0, sizeof a);
        a.np = 2; a.scale = 1.0f; a.relu = 1;
        a.M = E; a.N = H; a.ldc = H;
        a.p[0] = mkpair(h3s_h, H, w_g1h,     2 * H, H, nullptr, ei_st);      // src rows
        a.p[1] = mkpair(h3t_h, H, w_g1h + H, 2 * H, H, nullptr, ei_st + E);  // tgt cols
        a.bias0 = gate_b1;
        a.gatevec = gate_w2; a.gatebias = gate_b2; a.gateout = gate;
        launch_gemm(a, K_GATE, stream);
    }
    combine_kernel<<<ew, 256, 0, stream>>>(h3s_h, h3t_h, ei_st, gate, bufE1h, E);

    // mlp1: relu([combined | edge_attr] @ w1^T + b1)  -> bufE2h [E x 128] f16
    {
        GemmArgs a; memset(&a, 0, sizeof a);
        a.np = 2; a.scale = 1.0f; a.relu = 1;
        a.M = E; a.N = H; a.ldc = H; a.C = bufE2h;
        a.p[0] = mkpair(bufE1h, H, w_m1h, H + ED, H);
        a.p[1] = mkpair(ea_h, ED, w_m1h + H, H + ED, ED);
        a.bias0 = mlp_b1;
        launch_gemm(a, K_MLP128, stream);
    }
    // mlp2: relu(h1 @ w2^T + b2) -> bufE1h [E x 64] f16
    {
        GemmArgs a; memset(&a, 0, sizeof a);
        a.np = 1; a.scale = 1.0f; a.relu = 1;
        a.M = E; a.N = 64; a.ldc = 64; a.C = bufE1h;
        a.p[0] = mkpair(bufE2h, H, w_m2h, H, H);
        a.bias0 = mlp_b2;
        launch_gemm(a, K_MLP64, stream);
    }
    // mlp3: h2 @ w3_pad^T + b3 -> d_out [E x 2] f32 (W rows padded to 16)
    {
        GemmArgs a; memset(&a, 0, sizeof a);
        a.np = 1; a.scale = 1.0f; a.relu = 0;
        a.M = E; a.N = 2; a.ldc = 2; a.C = d_out;
        a.p[0] = mkpair(bufE1h, 64, w_m3h, 64, 64);
        a.bias0 = mlp_b3;
        launch_gemm(a, K_MLP2OUT, stream);
    }
}
